// SpatialGRU_7232724926625
// MI455X (gfx1250) — matrix operand order repaired and checked by reading, not compiled, not run on hardware
//
#include <hip/hip_runtime.h>
#include <hip/hip_bf16.h>
#include <math.h>

// ---------------------------------------------------------------------------
// SpatialGRU for MI455X (gfx1250, wave32, WMMA f16 16x16x32).
//
// Per cell (i,j):
//   GEMM1: A = [h_top|h_left|h_diag|x|0]  (64 x 224, f16 in LDS)
//          B = Wc packed weights          (512 rows x 224 K, f16, L2)
//          D (64 x 512 f32) = [rr_pre(192) | z_pre(256) | sw(64)]
//   epilogue: rr = sigmoid(rr_pre+b); A2 = rr * hu (f16, LDS)
//             z  = softmax over 4 gate blocks (f32, LDS); sw += b (f32, LDS)
//   GEMM2: h_new = tanh(sw + A2 @ Uc^T)   (64 x 64)
//   h = zl*h_left + zt*h_top + zd*h_diag + zi*h_new -> diagonal ring buffer
//
// 127 diagonal launches resolve the 2D wavefront dependency.
// ---------------------------------------------------------------------------

typedef __attribute__((ext_vector_type(16))) _Float16 v16h;
typedef __attribute__((ext_vector_type(8)))  _Float16 v8h;
typedef __attribute__((ext_vector_type(8)))  float    v8f;

#define KQ   224          // padded K for GEMM1 (208 real + 16 pad)
#define K2   192          // K for GEMM2 (exactly 6 * 32)
#define NQ   512          // GEMM1 output columns: 192 rr + 256 z + 64 sw

// workspace byte offsets
#define WS_WC    0                          // 512*224 f16  = 229376 B
#define WS_UC    229376                     //  64*224 f16  =  28672 B
#define WS_BIAS  258048                     // 512 f32      =   2048 B
#define WS_HRING 262144                     // 3*64*64*64 f32 = 3 MB

// LDS byte offsets (dynamic shared)
#define LDS_AQ   0                          // 64*224 f16 = 28672
#define LDS_A2   28672                      // 64*192 f16 = 24576
#define LDS_SW   53248                      // 64*64 f32  = 16384
#define LDS_Z    69632                      // 4*64*64 f32 = 65536
#define LDS_TOT  135168

__device__ __forceinline__ v16h cat16(v8h lo, v8h hi) {
    return __builtin_shufflevector(lo, hi,
        0,1,2,3,4,5,6,7,8,9,10,11,12,13,14,15);
}

// B fragment: lane = column N (lane&15), K = (lane>>4)*16 + 0..15 contiguous.
// W is row-major [N][KQ] f16, so each lane reads 32 contiguous bytes.
__device__ __forceinline__ v16h load_bfrag(const _Float16* __restrict__ W,
                                           int nrow, int koff) {
    const v8h* lo = (const v8h*)(W + (size_t)nrow * KQ + koff);
    const v8h* hi = (const v8h*)(W + (size_t)nrow * KQ + koff + 8);
    return cat16(*lo, *hi);
}

// A fragment from LDS (stride in halves): lane = row M (lane&15),
// K = k*32 + kh*16 + 0..15 contiguous.
__device__ __forceinline__ v16h load_afrag(const _Float16* A, int stride,
                                           int row, int kbase, int kh) {
    const v8h* lo = (const v8h*)(A + row * stride + kbase + kh * 16);
    const v8h* hi = (const v8h*)(A + row * stride + kbase + kh * 16 + 8);
    return cat16(*lo, *hi);
}

__device__ __forceinline__ v8f wmma_f16(v16h a, v16h b, v8f c) {
    return __builtin_amdgcn_wmma_f32_16x16x32_f16(
        false, a, false, b, (short)0, c, false, false);
}

// ------------------------- weight packing kernel ---------------------------
__global__ void sgru_prep(const float* __restrict__ wr_w,
                          const float* __restrict__ wr_b,
                          const float* __restrict__ wz_w,
                          const float* __restrict__ wz_b,
                          const float* __restrict__ wij_w,
                          const float* __restrict__ wij_b,
                          const float* __restrict__ U_w,
                          _Float16* __restrict__ Wc,
                          _Float16* __restrict__ Uc,
                          float* __restrict__ bias) {
    int idx = blockIdx.x * blockDim.x + threadIdx.x;
    int stride = gridDim.x * blockDim.x;
    for (int t = idx; t < NQ * KQ; t += stride) {
        int r = t / KQ, k = t % KQ;
        float v = 0.0f;
        if (r < 192)      { if (k < 208) v = wr_w[r * 208 + k]; }
        else if (r < 448) { if (k < 208) v = wz_w[(r - 192) * 208 + k]; }
        else              { if (k >= 192 && k < 208) v = wij_w[(r - 448) * 16 + (k - 192)]; }
        Wc[t] = (_Float16)v;
    }
    for (int t = idx; t < 64 * KQ; t += stride) {
        int r = t / KQ, k = t % KQ;
        Uc[t] = (_Float16)((k < 192) ? U_w[r * 192 + k] : 0.0f);
    }
    for (int t = idx; t < 512; t += stride) {
        bias[t] = (t < 192) ? wr_b[t] : ((t < 448) ? wz_b[t - 192] : wij_b[t - 448]);
    }
}

// ---------------------------- diagonal kernel ------------------------------
__global__ __launch_bounds__(256)
void sgru_diag(const float* __restrict__ inputs,
               const _Float16* __restrict__ Wc,
               const _Float16* __restrict__ Uc,
               const float* __restrict__ bias,
               float* __restrict__ hring,
               float* __restrict__ out,
               int d) {
    extern __shared__ char smem[];
    _Float16* Aq  = (_Float16*)(smem + LDS_AQ);   // 64 x 224
    _Float16* A2  = (_Float16*)(smem + LDS_A2);   // 64 x 192
    float*    swv = (float*)   (smem + LDS_SW);   // 64 x 64
    float*    zsm = (float*)   (smem + LDS_Z);    // 4 x 64 x 64

    const int i0 = (d > 63) ? (d - 63) : 0;
    const int i  = i0 + blockIdx.x;
    const int j  = d - i;
    const int tid = threadIdx.x;

    // ---------------- phase 0: stage A matrix into LDS --------------------
    const float* srcTop  = (i > 0)          ? hring + (((size_t)((d + 2) % 3) * 64 + j)     * 4096) : nullptr;
    const float* srcLeft = (j > 0)          ? hring + (((size_t)((d + 2) % 3) * 64 + (j-1)) * 4096) : nullptr;
    const float* srcDiag = (i > 0 && j > 0) ? hring + (((size_t)((d + 1) % 3) * 64 + (j-1)) * 4096) : nullptr;

    for (int t = tid; t < 4096; t += 256) {
        int b = t >> 6, u = t & 63;
        float ht = srcTop  ? srcTop[t]  : 0.0f;
        float hl = srcLeft ? srcLeft[t] : 0.0f;
        float hd = srcDiag ? srcDiag[t] : 0.0f;
        Aq[b * KQ + u]        = (_Float16)ht;   // hq block 0 = h_top
        Aq[b * KQ + 64 + u]   = (_Float16)hl;   // hq block 1 = h_left
        Aq[b * KQ + 128 + u]  = (_Float16)hd;   // hq block 2 = h_diag
    }
    for (int t = tid; t < 1024; t += 256) {     // x block (B x C)
        int b = t >> 4, c = t & 15;
        Aq[b * KQ + 192 + c] = (_Float16)inputs[(size_t)(b * 16 + c) * 4096 + i * 64 + j];
    }
    for (int t = tid; t < 1024; t += 256) {     // K pad 208..223
        int b = t >> 4, c = t & 15;
        Aq[b * KQ + 208 + c] = (_Float16)0.0f;
    }
    __syncthreads();

    // ---------------- phase 1: GEMM1 (64 x 512 x 224) ----------------------
    const int wave = tid >> 5, lane = tid & 31;
    const int mt   = wave & 3;                  // M tile 0..3
    const int row  = mt * 16 + (lane & 15);     // A row this lane supplies
    const int kh   = lane >> 4;                 // K half-select

    v16h afrag[7];
#pragma unroll
    for (int k = 0; k < 7; ++k)
        afrag[k] = load_afrag(Aq, KQ, row, k * 32, kh);

    if (wave < 4) {
        // rr tiles: N tiles 0..11  -> A2 = sigmoid(rr_pre + b) * hu
#pragma unroll 1
        for (int nt = 0; nt < 12; ++nt) {
            int n = nt * 16 + (lane & 15);
            v8f acc = {};
#pragma unroll
            for (int k = 0; k < 7; ++k)
                acc = wmma_f16(afrag[k], load_bfrag(Wc, n, k * 32 + kh * 16), acc);
            float bv = bias[n];
            // hu = [h_left | h_top | h_diag]; swap first two 64-blocks of hq
            int hucol = (n < 64) ? (64 + n) : ((n < 128) ? (n - 64) : n);
#pragma unroll
            for (int v = 0; v < 8; ++v) {
                int m = mt * 16 + v + kh * 8;
                float r  = 1.0f / (1.0f + __expf(-(acc[v] + bv)));
                float hu = (float)Aq[m * KQ + hucol];
                A2[m * K2 + n] = (_Float16)(r * hu);
            }
        }
        // sw tiles: N tiles 28..31 -> swv = sw_pre + b
#pragma unroll 1
        for (int nt = 28; nt < 32; ++nt) {
            int n = nt * 16 + (lane & 15);
            v8f acc = {};
#pragma unroll
            for (int k = 0; k < 7; ++k)
                acc = wmma_f16(afrag[k], load_bfrag(Wc, n, k * 32 + kh * 16), acc);
            float bv = bias[n];
#pragma unroll
            for (int v = 0; v < 8; ++v) {
                int m = mt * 16 + v + kh * 8;
                swv[m * 64 + (n - 448)] = acc[v] + bv;
            }
        }
    } else {
        // z tiles 12..27; coupled gate quads {12+q, 16+q, 20+q, 24+q}
#pragma unroll 1
        for (int q = 0; q < 4; ++q) {
            v8f acc[4] = {{}, {}, {}, {}};
#pragma unroll
            for (int g = 0; g < 4; ++g) {
                int n = (12 + q + 4 * g) * 16 + (lane & 15);
#pragma unroll
                for (int k = 0; k < 7; ++k)
                    acc[g] = wmma_f16(afrag[k], load_bfrag(Wc, n, k * 32 + kh * 16), acc[g]);
                float bv = bias[n];
#pragma unroll
                for (int v = 0; v < 8; ++v) acc[g][v] += bv;
            }
            int u = q * 16 + (lane & 15);
#pragma unroll
            for (int v = 0; v < 8; ++v) {
                int m = mt * 16 + v + kh * 8;
                float z0 = acc[0][v], z1 = acc[1][v], z2 = acc[2][v], z3 = acc[3][v];
                float mx = fmaxf(fmaxf(z0, z1), fmaxf(z2, z3));
                float e0 = __expf(z0 - mx), e1 = __expf(z1 - mx);
                float e2 = __expf(z2 - mx), e3 = __expf(z3 - mx);
                float inv = 1.0f / (e0 + e1 + e2 + e3);
                zsm[(0 * 64 + m) * 64 + u] = e0 * inv;  // zi
                zsm[(1 * 64 + m) * 64 + u] = e1 * inv;  // zl
                zsm[(2 * 64 + m) * 64 + u] = e2 * inv;  // zt
                zsm[(3 * 64 + m) * 64 + u] = e3 * inv;  // zd
            }
        }
    }
    __syncthreads();

    // ---------------- phase 2: GEMM2 (64 x 64 x 192) + combine -------------
    float* hout = hring + (((size_t)(d % 3) * 64 + j) * 4096);
    const int ntb = (wave >> 2) * 2;
#pragma unroll 1
    for (int nt = ntb; nt < ntb + 2; ++nt) {
        int n = nt * 16 + (lane & 15);
        v8f acc = {};
#pragma unroll
        for (int k = 0; k < 6; ++k) {
            v16h a = load_afrag(A2, K2, row, k * 32, kh);
            v16h b = load_bfrag(Uc, n, k * 32 + kh * 16);
            acc = wmma_f16(a, b, acc);
        }
#pragma unroll
        for (int v = 0; v < 8; ++v) {
            int m = mt * 16 + v + kh * 8;
            float hnew = tanhf(acc[v] + swv[m * 64 + n]);
            float zi = zsm[(0 * 64 + m) * 64 + n];
            float zl = zsm[(1 * 64 + m) * 64 + n];
            float zt = zsm[(2 * 64 + m) * 64 + n];
            float zd = zsm[(3 * 64 + m) * 64 + n];
            float hl = (float)Aq[m * KQ + 64 + n];
            float ht = (float)Aq[m * KQ + n];
            float hd = (float)Aq[m * KQ + 128 + n];
            float h  = zl * hl + zt * ht + zd * hd + zi * hnew;
            hout[m * 64 + n] = h;
            if (i == 63 && j == 63) out[m * 64 + n] = h;
        }
    }
}

// ------------------------------- launcher ----------------------------------
extern "C" void kernel_launch(void* const* d_in, const int* in_sizes, int n_in,
                              void* d_out, int out_size, void* d_ws, size_t ws_size,
                              hipStream_t stream) {
    const float* inputs = (const float*)d_in[0];
    const float* wr_w   = (const float*)d_in[1];
    const float* wr_b   = (const float*)d_in[2];
    const float* wz_w   = (const float*)d_in[3];
    const float* wz_b   = (const float*)d_in[4];
    const float* wij_w  = (const float*)d_in[5];
    const float* wij_b  = (const float*)d_in[6];
    const float* U_w    = (const float*)d_in[7];

    char* ws = (char*)d_ws;
    _Float16* Wc   = (_Float16*)(ws + WS_WC);
    _Float16* Uc   = (_Float16*)(ws + WS_UC);
    float*    bias = (float*)   (ws + WS_BIAS);
    float*    hring= (float*)   (ws + WS_HRING);

    sgru_prep<<<112, 256, 0, stream>>>(wr_w, wr_b, wz_w, wz_b, wij_w, wij_b,
                                       U_w, Wc, Uc, bias);

    for (int d = 0; d < 127; ++d) {
        int ilo = (d > 63) ? (d - 63) : 0;
        int ihi = (d < 63) ? d : 63;
        int ncells = ihi - ilo + 1;
        sgru_diag<<<ncells, 256, LDS_TOT, stream>>>(inputs, Wc, Uc, bias, hring,
                                                    (float*)d_out, d);
    }
}
